// SparseSamplingAttention_10436770529679
// MI455X (gfx1250) — compile-verified, matrix operand order checked by reading, not run on hardware
//
#include <hip/hip_runtime.h>
#include <hip/hip_bf16.h>
#include <math.h>

// Problem constants (match reference)
#define Bv     8
#define DIMv   192
#define HEADSv 6
#define WSv    8
#define Hv     128
#define Wv     128
#define HDv    32
#define HWv    (Hv * Wv)          // 16384
#define NPRED  (Bv * HEADSv)      // 48
#define NWIN   16                 // windows per side
#define TOPKv  13107              // int(0.8 * 16384)
#define NPIX   (Bv * HWv)         // 131072 total pixels

typedef __attribute__((ext_vector_type(16))) __bf16          v16bf;
typedef __attribute__((ext_vector_type(8)))  float           v8f;
typedef __attribute__((ext_vector_type(4)))  float           v4f;
typedef __attribute__((ext_vector_type(8)))  unsigned short  us8;
typedef __attribute__((ext_vector_type(4)))  unsigned int    v4u;
typedef __attribute__((ext_vector_type(4)))  int             v4i;
typedef __attribute__((ext_vector_type(8)))  int             v8i;

// Native bf16 conversions (RNE fptrunc -> hardware cvt on gfx1250)
static __device__ __forceinline__ unsigned short f2bfu(float f) {
    return __builtin_bit_cast(unsigned short, (__bf16)f);
}
static __device__ __forceinline__ float bf2f(unsigned short s) {
    return (float)__builtin_bit_cast(__bf16, s);
}
static __device__ __forceinline__ __bf16 usbf(unsigned short s) {
    return __builtin_bit_cast(__bf16, s);
}

static __device__ __forceinline__ unsigned ordkey(float f) {
    unsigned u = __float_as_uint(f);
    return u ^ ((u >> 31) ? 0xFFFFFFFFu : 0x80000000u);
}

// Load a 16-element bf16 WMMA A/B fragment from an LDS row where the K values
// of this lane live at [off0, off0+8) and [off0+16, off0+24)  (ISA 7.12.2
// 16-bit operand layout: elems 0-7 -> K = hi*8+e, elems 8-15 -> K = 16+hi*8+e).
static __device__ __forceinline__ v16bf ldfrag_lds(const unsigned short* row, int off0) {
    us8 a = *(const us8*)(row + off0);
    us8 b = *(const us8*)(row + off0 + 16);
    v16bf r;
#pragma unroll
    for (int e = 0; e < 8; ++e) { r[e] = usbf(a[e]); r[e + 8] = usbf(b[e]); }
    return r;
}
// Same fragment from an f32 global row (weights), converting to bf16.
static __device__ __forceinline__ v16bf ldfrag_w(const float* row, int off0) {
    v4f w0 = *(const v4f*)(row + off0);
    v4f w1 = *(const v4f*)(row + off0 + 4);
    v4f w2 = *(const v4f*)(row + off0 + 16);
    v4f w3 = *(const v4f*)(row + off0 + 20);
    v16bf r;
#pragma unroll
    for (int e = 0; e < 4; ++e) {
        r[e]      = (__bf16)w0[e];
        r[e + 4]  = (__bf16)w1[e];
        r[e + 8]  = (__bf16)w2[e];
        r[e + 12] = (__bf16)w3[e];
    }
    return r;
}

// ---------------------------------------------------------------------------
// TDM: DMA a 2-D tile (32 contiguous elems/row x `rows` rows, row stride
// `stride` elems, element size 2^dsz bytes) from global into LDS at byte
// offset lds_off.  D# packing per CDNA5 ISA §8.3/8.4 (count=1, type=2,
// no pad/iterate/multicast).
// ---------------------------------------------------------------------------
#if __has_builtin(__builtin_amdgcn_tensor_load_to_lds)
#define HAVE_TDM 1
static __device__ __forceinline__ void tdm_load_tile32(const void* gsrc, unsigned lds_off,
                                                       int rows, unsigned stride, int dsz) {
    unsigned long long ga = (unsigned long long)(uintptr_t)gsrc;
    v4u g0;
    g0[0] = 1u;                                         // count=1, user descriptor
    g0[1] = lds_off;                                    // lds_addr (bytes)
    g0[2] = (unsigned)(ga & 0xFFFFFFFFu);               // global_addr[31:0]
    g0[3] = (unsigned)((ga >> 32) & 0x01FFFFFFu) | 0x80000000u; // addr[56:32] | type=2
    v8i g1;
    g1[0] = dsz << 16;                                  // data_size (0=1B,1=2B,2=4B)
    g1[1] = (int)0xFFFF0000u;                           // tensor_dim0 lo16 (dim0=0x7FFFFFFF)
    g1[2] = (int)(0x7FFFu | 0xFFFF0000u);               // tensor_dim0 hi16 | tensor_dim1 lo16
    g1[3] = (int)(0x7FFFu | (32u << 16));               // tensor_dim1 hi16 | tile_dim0=32
    g1[4] = rows;                                       // tile_dim1=rows, tile_dim2=0
    g1[5] = (int)stride;                                // tensor_dim0_stride[31:0]
    g1[6] = 0;                                          // stride0 hi | stride1 lo
    g1[7] = 0;                                          // stride1 hi
    v4i g2 = {0, 0, 0, 0};
    v4i g3 = {0, 0, 0, 0};
#if defined(__clang_major__) && __clang_major__ >= 23
    v8i g4 = {0, 0, 0, 0, 0, 0, 0, 0};
    __builtin_amdgcn_tensor_load_to_lds(g0, g1, g2, g3, g4, 0);
#else
    __builtin_amdgcn_tensor_load_to_lds(g0, g1, g2, g3, 0);
#endif
}
static __device__ __forceinline__ void tdm_wait() {
#if __has_builtin(__builtin_amdgcn_s_wait_tensorcnt)
    __builtin_amdgcn_s_wait_tensorcnt(0);
#else
    asm volatile("s_wait_tensorcnt 0x0" ::: "memory");
#endif
}
#else
#define HAVE_TDM 0
#endif

// ---------------------------------------------------------------------------
// K1: per-(b,c) exact k-th-largest threshold via 4-pass 256-bin radix select
// ---------------------------------------------------------------------------
__global__ __launch_bounds__(256) void topk_thresh_kernel(const float* __restrict__ var,
                                                          unsigned int* __restrict__ thr) {
    int row = blockIdx.x;                       // b*DIM + c  (1536 rows)
    const float* v = var + (size_t)row * HWv;
    __shared__ unsigned int hist[256];
    __shared__ unsigned int s_prefix;
    __shared__ unsigned int s_remaining;
    if (threadIdx.x == 0) { s_prefix = 0u; s_remaining = TOPKv; }
    __syncthreads();
    for (int pass = 0; pass < 4; ++pass) {
        int shift = 24 - pass * 8;
        unsigned hs = (unsigned)(shift + 8);
        unsigned himask = (hs >= 32u) ? 0u : (0xFFFFFFFFu << hs);
        hist[threadIdx.x] = 0u;
        __syncthreads();
        unsigned prefix = s_prefix;
        for (int i = threadIdx.x; i < HWv; i += 256) {
            unsigned key = ordkey(v[i]);
            if ((key & himask) == (prefix & himask))
                atomicAdd(&hist[(key >> shift) & 255u], 1u);
        }
        __syncthreads();
        if (threadIdx.x == 0) {
            unsigned rem = s_remaining;
            for (int bin = 255; bin >= 0; --bin) {
                unsigned c = hist[bin];
                if (c >= rem) { s_prefix = prefix | ((unsigned)bin << shift); break; }
                rem -= c;
            }
            s_remaining = rem;
        }
        __syncthreads();
    }
    if (threadIdx.x == 0) thr[row] = s_prefix;
}

// ---------------------------------------------------------------------------
// K2: pooled = leaky_relu( avgpool8( x * where(key>=thr, 1.0, 0.6) ) )
// ---------------------------------------------------------------------------
__global__ __launch_bounds__(256) void pool_kernel(const float* __restrict__ x,
                                                   const float* __restrict__ var,
                                                   const unsigned int* __restrict__ thr,
                                                   float* __restrict__ pooled) {
    int gid = blockIdx.x * 256 + threadIdx.x;   // (b*DIM+c)*256 + (wy*16+wx)
    int pos = gid & 255;
    int row = gid >> 8;
    int wy = pos >> 4, wx = pos & 15;
    unsigned t = thr[row];
    const float* xb = x + (size_t)row * HWv;
    const float* vb = var + (size_t)row * HWv;
    float s = 0.0f;
    for (int i = 0; i < 8; ++i) {
        int y = wy * 8 + i;
        for (int j = 0; j < 8; ++j) {
            int off = y * Wv + wx * 8 + j;
            float w = (ordkey(vb[off]) >= t) ? 1.0f : 0.6f;
            s += xb[off] * w;
        }
    }
    s *= (1.0f / 64.0f);
    pooled[gid] = (s >= 0.0f) ? s : 0.01f * s;  // leaky_relu(0.01)
}

// ---------------------------------------------------------------------------
// K3: biases (÷16 folded) and scales per (n_idx, ch, wy, wx)
// ---------------------------------------------------------------------------
__global__ __launch_bounds__(256) void predictor_kernel(const float* __restrict__ pooled,
                                                        const float* __restrict__ bias_w,
                                                        const float* __restrict__ bias_b,
                                                        const float* __restrict__ scale_w,
                                                        const float* __restrict__ scale_b,
                                                        float* __restrict__ biases,
                                                        float* __restrict__ scales) {
    int gid = blockIdx.x * 256 + threadIdx.x;   // 48*2*256 = 24576
    if (gid >= NPRED * 2 * 256) return;
    int pos = gid & 255;
    int rest = gid >> 8;                        // n_idx*2 + ch
    int ch = rest & 1;
    int n  = rest >> 1;                         // b*HEADS + head
    int batch = n / HEADSv;
    int head  = n % HEADSv;
    int o = head * 2 + ch;
    const float* pb = pooled + (size_t)batch * DIMv * 256 + pos;
    float sb = 0.0f, ss = 0.0f;
    for (int c = 0; c < DIMv; ++c) {
        float p = pb[(size_t)c * 256];
        sb += p * bias_w[o * DIMv + c];
        ss += p * scale_w[o * DIMv + c];
    }
    sb += bias_b[o];
    ss += scale_b[o];
    biases[gid] = sb * (1.0f / 16.0f);
    scales[gid] = ss;
}

// ---------------------------------------------------------------------------
// K4: QKV GEMM, bf16 WMMA 16x16x32, M=131072 N=576 K=192.
// Block = 32 pixels, TDM-staged f32 tile -> bf16 As[p][c]; 8 waves (2M x 4N)
// cover all 36 N-tiles so x is read exactly once.  Outputs stored as bf16.
// ---------------------------------------------------------------------------
__global__ __launch_bounds__(256) void qkv_gemm_kernel(const float* __restrict__ x,
                                                       const float* __restrict__ qkv_w,
                                                       const float* __restrict__ qkv_b,
                                                       unsigned short* __restrict__ qbuf,
                                                       unsigned short* __restrict__ kbuf,
                                                       unsigned short* __restrict__ vbuf) {
    __shared__ __align__(16) unsigned short As[32][DIMv];   // bf16, [pixel][channel]
    int tid = threadIdx.x;
    int p0 = blockIdx.x * 32;                   // 32 pixels per block
    int batch = p0 >> 14;
    int s0 = p0 & (HWv - 1);
    const float* gsrc = x + (size_t)batch * DIMv * HWv + s0;

#if HAVE_TDM
    __shared__ __align__(16) float xf[DIMv][32];            // f32 staging (TDM dest)
    if (tid < 32) {
        tdm_load_tile32(gsrc, (unsigned)(uintptr_t)&xf[0][0], DIMv, HWv, 2);
        tdm_wait();
    }
    __syncthreads();
    for (int idx = tid; idx < DIMv * 32; idx += 256) {
        int c = idx >> 5, p = idx & 31;
        As[p][c] = f2bfu(xf[c][p]);
    }
#else
    for (int idx = tid; idx < DIMv * 32; idx += 256) {
        int c = idx >> 5, p = idx & 31;
        As[p][c] = f2bfu(gsrc[(size_t)c * HWv + p]);
    }
#endif
    __syncthreads();

    int lane = tid & 31;
    int wave = tid >> 5;
    int lo = lane & 15;
    int hi = lane >> 4;
    int msub = wave >> 2;                       // 0..1
    int nsub = wave & 3;                        // 0..3
    const unsigned short* arow = &As[msub * 16 + lo][0];

    for (int ng = 0; ng < 9; ++ng) {            // 9 groups x 4 = 36 N-tiles
        int ntile = ng * 4 + nsub;
        int o = ntile * 16 + lo;
        const float* wrow = qkv_w + (size_t)o * DIMv;
        if (ntile + 4 < 36) __builtin_prefetch(qkv_w + (size_t)(o + 64) * DIMv, 0, 1);
        v8f acc = {};
#pragma unroll
        for (int k0 = 0; k0 < 6; ++k0) {
            int cb = k0 * 32;
            v16bf a  = ldfrag_lds(arow, cb + hi * 8);
            v16bf bm = ldfrag_w(wrow, cb + hi * 8);
            acc = __builtin_amdgcn_wmma_f32_16x16x32_bf16(false, a, false, bm,
                                                          (short)0, acc, false, false);
        }
        float bias = qkv_b[o];
        int part = o / DIMv;                    // 0=q 1=k 2=v
        int c = o - part * DIMv;                // plane channel (head*32+hd)
        unsigned short* dst = (part == 0) ? qbuf : ((part == 1) ? kbuf : vbuf);
#pragma unroll
        for (int r = 0; r < 8; ++r) {
            int pp = p0 + msub * 16 + r + 8 * hi;
            dst[((size_t)batch * DIMv + c) * HWv + (pp & (HWv - 1))] = f2bfu(acc[r] + bias);
        }
    }
}

// ---------------------------------------------------------------------------
// K5: fused grid-sample + window attention.  One block per (window, head).
// q/k/v planes are bf16; all WMMA fragments load as 16-byte LDS vectors.
// ---------------------------------------------------------------------------
__global__ __launch_bounds__(128) void attn_kernel(const unsigned short* __restrict__ qbuf,
                                                   const unsigned short* __restrict__ kbuf,
                                                   const unsigned short* __restrict__ vbuf,
                                                   const float* __restrict__ biases,
                                                   const float* __restrict__ scales,
                                                   const float* __restrict__ rel_table,
                                                   unsigned short* __restrict__ abuf) {
    int win   = blockIdx.x;                     // b*256 + wy*16 + wx
    int head  = blockIdx.y;
    int batch = win >> 8;
    int wy    = (win >> 4) & 15;
    int wx    = win & 15;
    int n_idx = batch * HEADSv + head;
    int tid   = threadIdx.x;

    __shared__ __align__(16) unsigned short qs[64][32];     // [pix][hd]
    __shared__ __align__(16) unsigned short kss[64][32];    // [pix][hd]
    __shared__ __align__(16) unsigned short vssT[32][64];   // [hd][pix]  (transposed)
    __shared__ __align__(16) unsigned short attnS[64][64];  // [row][col]
    __shared__ float gxs[64], gys[64];

    int posw = wy * 16 + wx;
    float sx = scales[(n_idx * 2 + 0) * 256 + posw];
    float sy = scales[(n_idx * 2 + 1) * 256 + posw];
    float bx = biases[(n_idx * 2 + 0) * 256 + posw];
    float by = biases[(n_idx * 2 + 1) * 256 + posw];

    if (tid < 64) {
        int i = tid >> 3, j = tid & 7;
        gxs[tid] = wx * 8 + 3.5f + ((float)j - 3.5f) * (1.0f + sx) + bx * 63.5f;
        gys[tid] = wy * 8 + 3.5f + ((float)i - 3.5f) * (1.0f + sy) + by * 63.5f;
    }
    __syncthreads();

    const float HD_SCALE = 0.17677669529663687f; // 32^-0.5
    size_t chbase = ((size_t)batch * DIMv + head * HDv) * HWv;

    for (int idx = tid; idx < 64 * HDv; idx += 128) {
        int pix = idx >> 5, hd = idx & 31;
        int i = pix >> 3, j = pix & 7;
        int y = wy * 8 + i, xx = wx * 8 + j;
        size_t cb = chbase + (size_t)hd * HWv;
        qs[pix][hd] = f2bfu(bf2f(qbuf[cb + y * Wv + xx]) * HD_SCALE);

        float gx = gxs[pix], gy = gys[pix];
        float x0f = floorf(gx), y0f = floorf(gy);
        int x0 = (int)x0f, y0 = (int)y0f;
        float fx = gx - x0f, fy = gy - y0f;
        float kacc = 0.0f, vacc = 0.0f;
#pragma unroll
        for (int dy = 0; dy < 2; ++dy)
#pragma unroll
            for (int dx = 0; dx < 2; ++dx) {
                int yy = y0 + dy, xc = x0 + dx;
                float w = (dx ? fx : 1.0f - fx) * (dy ? fy : 1.0f - fy);
                if (yy >= 0 && yy < Hv && xc >= 0 && xc < Wv) {
                    int off = yy * Wv + xc;
                    kacc += w * bf2f(kbuf[cb + off]);
                    vacc += w * bf2f(vbuf[cb + off]);
                }
            }
        kss[pix][hd]  = f2bfu(kacc);
        vssT[hd][pix] = f2bfu(vacc);
    }
    __syncthreads();

    int lane = tid & 31, wave = tid >> 5;
    int hi = lane >> 4, lo = lane & 15;
    int mbase = wave * 16;

    // ---- scores = q . k^T  (16 rows per wave, N=64, K=32) ----
    v16bf aq = ldfrag_lds(&qs[mbase + lo][0], hi * 8);
    v8f sc[4];
#pragma unroll
    for (int t = 0; t < 4; ++t) {
        v16bf bk = ldfrag_lds(&kss[t * 16 + lo][0], hi * 8);
        v8f z = {};
        sc[t] = __builtin_amdgcn_wmma_f32_16x16x32_bf16(false, aq, false, bk,
                                                        (short)0, z, false, false);
    }

    // ---- rel bias + row softmax (rows live in 16-lane halves) ----
#pragma unroll
    for (int r = 0; r < 8; ++r) {
        int m = mbase + r + 8 * hi;             // window-local row 0..63
        float mx = -3.0e38f;
#pragma unroll
        for (int t = 0; t < 4; ++t) {
            int nn = t * 16 + lo;
            int dh = (m >> 3) - (nn >> 3) + 7;
            int dw = (m & 7) - (nn & 7) + 7;
            float v = sc[t][r] + rel_table[(dh * (2 * WSv - 1) + dw) * HEADSv + head];
            sc[t][r] = v;
            mx = fmaxf(mx, v);
        }
        for (int msk = 1; msk < 16; msk <<= 1) mx = fmaxf(mx, __shfl_xor(mx, msk, 32));
        float sm = 0.0f;
#pragma unroll
        for (int t = 0; t < 4; ++t) {
            float e = __expf(sc[t][r] - mx);
            sc[t][r] = e;
            sm += e;
        }
        for (int msk = 1; msk < 16; msk <<= 1) sm += __shfl_xor(sm, msk, 32);
        float inv = 1.0f / sm;
#pragma unroll
        for (int t = 0; t < 4; ++t) attnS[m][t * 16 + lo] = f2bfu(sc[t][r] * inv);
    }
    __syncthreads();

    // ---- out = attn . v  (M=16, N=32, K=64) ----
    v8f oacc[2];
    oacc[0] = (v8f){};
    oacc[1] = (v8f){};
#pragma unroll
    for (int k2 = 0; k2 < 2; ++k2) {
        v16bf aA = ldfrag_lds(&attnS[mbase + lo][0], k2 * 32 + hi * 8);
#pragma unroll
        for (int nt = 0; nt < 2; ++nt) {
            v16bf bV = ldfrag_lds(&vssT[nt * 16 + lo][0], k2 * 32 + hi * 8);
            oacc[nt] = __builtin_amdgcn_wmma_f32_16x16x32_bf16(false, aA, false, bV,
                                                               (short)0, oacc[nt],
                                                               false, false);
        }
    }
#pragma unroll
    for (int nt = 0; nt < 2; ++nt)
#pragma unroll
        for (int r = 0; r < 8; ++r) {
            int m = mbase + r + 8 * hi;
            int hd = nt * 16 + lo;
            int i = m >> 3, j = m & 7;
            int y = wy * 8 + i, xx = wx * 8 + j;
            abuf[chbase + (size_t)hd * HWv + y * Wv + xx] = f2bfu(oacc[nt][r]);
        }
}

// ---------------------------------------------------------------------------
// K6: output projection GEMM, M=131072 N=192 K=192, bf16 A-plane (TDM-staged
// with 2-byte elements, conversion-free), writes f32 d_out.
// ---------------------------------------------------------------------------
__global__ __launch_bounds__(256) void proj_gemm_kernel(const unsigned short* __restrict__ abuf,
                                                        const float* __restrict__ proj_w,
                                                        const float* __restrict__ proj_b,
                                                        float* __restrict__ out) {
    __shared__ __align__(16) unsigned short As[32][DIMv];
    int tid = threadIdx.x;
    int p0 = blockIdx.x * 32;
    int batch = p0 >> 14;
    int s0 = p0 & (HWv - 1);
    const unsigned short* gsrc = abuf + (size_t)batch * DIMv * HWv + s0;

#if HAVE_TDM
    __shared__ __align__(16) unsigned short xh[DIMv][32];   // bf16 staging (TDM dest)
    if (tid < 32) {
        tdm_load_tile32(gsrc, (unsigned)(uintptr_t)&xh[0][0], DIMv, HWv, 1);
        tdm_wait();
    }
    __syncthreads();
    for (int idx = tid; idx < DIMv * 32; idx += 256) {
        int c = idx >> 5, p = idx & 31;
        As[p][c] = xh[c][p];                    // pure transpose, no convert
    }
#else
    for (int idx = tid; idx < DIMv * 32; idx += 256) {
        int c = idx >> 5, p = idx & 31;
        As[p][c] = gsrc[(size_t)c * HWv + p];
    }
#endif
    __syncthreads();

    int lane = tid & 31;
    int wave = tid >> 5;
    int lo = lane & 15;
    int hi = lane >> 4;
    int msub = wave >> 2;
    int nsub = wave & 3;
    const unsigned short* arow = &As[msub * 16 + lo][0];

    for (int ng = 0; ng < 3; ++ng) {            // 3 groups x 4 = 12 N-tiles
        int ntile = ng * 4 + nsub;
        int o = ntile * 16 + lo;
        const float* wrow = proj_w + (size_t)o * DIMv;
        if (ntile + 4 < 12) __builtin_prefetch(proj_w + (size_t)(o + 64) * DIMv, 0, 1);
        v8f acc = {};
#pragma unroll
        for (int k0 = 0; k0 < 6; ++k0) {
            int cb = k0 * 32;
            v16bf a  = ldfrag_lds(arow, cb + hi * 8);
            v16bf bm = ldfrag_w(wrow, cb + hi * 8);
            acc = __builtin_amdgcn_wmma_f32_16x16x32_bf16(false, a, false, bm,
                                                          (short)0, acc, false, false);
        }
        float bias = proj_b[o];
#pragma unroll
        for (int r = 0; r < 8; ++r) {
            int pp = p0 + msub * 16 + r + 8 * hi;
            out[((size_t)batch * DIMv + o) * HWv + (pp & (HWv - 1))] = acc[r] + bias;
        }
    }
}

// ---------------------------------------------------------------------------
extern "C" void kernel_launch(void* const* d_in, const int* in_sizes, int n_in,
                              void* d_out, int out_size, void* d_ws, size_t ws_size,
                              hipStream_t stream) {
    (void)in_sizes; (void)n_in; (void)out_size; (void)ws_size;
    const float* x        = (const float*)d_in[0];
    const float* var      = (const float*)d_in[1];
    const float* qkv_w    = (const float*)d_in[2];
    const float* qkv_b    = (const float*)d_in[3];
    const float* bias_w   = (const float*)d_in[4];
    const float* bias_b   = (const float*)d_in[5];
    const float* scale_w  = (const float*)d_in[6];
    const float* scale_b  = (const float*)d_in[7];
    const float* proj_w   = (const float*)d_in[8];
    const float* proj_b   = (const float*)d_in[9];
    const float* rel_tab  = (const float*)d_in[10];
    float* out = (float*)d_out;

    char* ws = (char*)d_ws;
    size_t off = 0;
    auto take = [&](size_t bytes) { void* p = ws + off; off = (off + bytes + 255) & ~(size_t)255; return p; };
    unsigned int* thr    = (unsigned int*)take((size_t)Bv * DIMv * 4);
    float* pooled        = (float*)take((size_t)Bv * DIMv * 256 * 4);
    float* biases        = (float*)take((size_t)NPRED * 2 * 256 * 4);
    float* scales        = (float*)take((size_t)NPRED * 2 * 256 * 4);
    unsigned short* qbuf = (unsigned short*)take((size_t)Bv * DIMv * HWv * 2);  // bf16
    unsigned short* kbuf = (unsigned short*)take((size_t)Bv * DIMv * HWv * 2);  // bf16
    unsigned short* vbuf = (unsigned short*)take((size_t)Bv * DIMv * HWv * 2);  // bf16
    unsigned short* abuf = (unsigned short*)take((size_t)Bv * DIMv * HWv * 2);  // bf16

    topk_thresh_kernel<<<Bv * DIMv, 256, 0, stream>>>(var, thr);
    pool_kernel<<<Bv * DIMv, 256, 0, stream>>>(x, var, thr, pooled);
    predictor_kernel<<<(NPRED * 2 * 256 + 255) / 256, 256, 0, stream>>>(
        pooled, bias_w, bias_b, scale_w, scale_b, biases, scales);
    qkv_gemm_kernel<<<dim3(NPIX / 32), 256, 0, stream>>>(
        x, qkv_w, qkv_b, qbuf, kbuf, vbuf);
    attn_kernel<<<dim3(Bv * NWIN * NWIN, HEADSv), 128, 0, stream>>>(
        qbuf, kbuf, vbuf, biases, scales, rel_tab, abuf);
    proj_gemm_kernel<<<dim3(NPIX / 32), 256, 0, stream>>>(
        abuf, proj_w, proj_b, out);
}
